// SimpleEdgeNet_1331439861969
// MI455X (gfx1250) — compile-verified, matrix operand order checked by reading, not run on hardware
//
#include <hip/hip_runtime.h>

typedef __attribute__((ext_vector_type(16))) __bf16 v16bf;
typedef __attribute__((ext_vector_type(8)))  float  v8f;

#define NN 50000
#define NE 800000
#define HD 64

// ---- helpers -------------------------------------------------------------

__device__ __forceinline__ unsigned short f2bf(float f) {
  return __builtin_bit_cast(unsigned short, (__bf16)f);
}
__device__ __forceinline__ __bf16 us2bf(unsigned short u) {
  return __builtin_bit_cast(__bf16, u);
}
__device__ __forceinline__ v8f wmma_bf16(v16bf a, v16bf b, v8f c) {
  // D = A(16x32 bf16) x B(32x16 bf16) + C(16x16 f32)
  return __builtin_amdgcn_wmma_f32_16x16x32_bf16(false, a, false, b, (short)0, c,
                                                 false, false);
}
// K index inside a 32-wide chunk for A/B fragment element j, lane group hi.
// lo lanes: K in {0..7, 16..23}; hi lanes: K in {8..15, 24..31}
__device__ __forceinline__ int kmap(int j, int hi) {
  return (j < 8 ? j : j + 8) + (hi << 3);
}

// ---- weight packing: f32 [K x N] row-major -> bf16 B-fragments -----------
// layout: frag f = t*nC + c ; element ((f*32 + lane)*16 + j)
__global__ void pack_weight_kernel(const float* __restrict__ W,
                                   unsigned short* __restrict__ out,
                                   int K, int N) {
  int idx = blockIdx.x * blockDim.x + threadIdx.x;
  int nC = K >> 5, nT = N >> 4;
  int total = nC * nT * 512;
  if (idx >= total) return;
  int j = idx & 15;
  int lane = (idx >> 4) & 31;
  int f = idx >> 9;
  int c = f % nC, t = f / nC;
  int hi = lane >> 4, m = lane & 15;
  int k = 32 * c + kmap(j, hi);
  int n = 16 * t + m;
  out[idx] = f2bf(W[k * N + n]);
}

// ---- scalar encoders (tiny FLOP count) -----------------------------------

__global__ void node_encoder_kernel(const float* __restrict__ x,
                                    const float* __restrict__ W0, const float* __restrict__ b0,
                                    const float* __restrict__ W1, const float* __restrict__ b1,
                                    float* __restrict__ h) {
  int i = blockIdx.x * blockDim.x + threadIdx.x;
  if (i >= NN) return;
  float in[6];
#pragma unroll
  for (int d = 0; d < 6; ++d) in[d] = x[i * 6 + d];
  float hid[HD];
  for (int o = 0; o < HD; ++o) {
    float s = b0[o];
#pragma unroll
    for (int d = 0; d < 6; ++d) s += in[d] * W0[d * HD + o];
    hid[o] = s > 0.f ? s : 0.f;
  }
  for (int o = 0; o < HD; ++o) {
    float s = b1[o];
#pragma unroll
    for (int k = 0; k < HD; ++k) s += hid[k] * W1[k * HD + o];
    h[i * HD + o] = s;
  }
}

__global__ void edge_encoder_kernel(const float* __restrict__ ea,
                                    const float* __restrict__ W0, const float* __restrict__ b0,
                                    const float* __restrict__ W1, const float* __restrict__ b1,
                                    float* __restrict__ e) {
  int i = blockIdx.x * blockDim.x + threadIdx.x;
  if (i >= NE) return;
  float in[8];
#pragma unroll
  for (int d = 0; d < 8; ++d) in[d] = ea[i * 8 + d];
  float hid[HD];
  for (int o = 0; o < HD; ++o) {
    float s = b0[o];
#pragma unroll
    for (int d = 0; d < 8; ++d) s += in[d] * W0[d * HD + o];
    hid[o] = s > 0.f ? s : 0.f;
  }
  for (int o = 0; o < HD; ++o) {
    float s = b1[o];
#pragma unroll
    for (int k = 0; k < HD; ++k) s += hid[k] * W1[k * HD + o];
    e[i * HD + o] = s;
  }
}

__global__ void zero_kernel(float* __restrict__ p, int n) {
  int i = blockIdx.x * blockDim.x + threadIdx.x;
  if (i < n) p[i] = 0.f;
}

// ---- edge update: e = MLP(192->128->64)(h[src]||h[dst]||e), agg += e -----

__global__ void __launch_bounds__(128)
edge_update_kernel(const float* __restrict__ h, float* __restrict__ e,
                   const int* __restrict__ edge_index, float* __restrict__ agg,
                   const unsigned short* __restrict__ W1p, const float* __restrict__ b1,
                   const unsigned short* __restrict__ W2p, const float* __restrict__ b2) {
  __shared__ unsigned short T[4][16 * 128];
  const int lane = threadIdx.x & 31;
  const int wave = threadIdx.x >> 5;
  const int tile = blockIdx.x * 4 + wave;           // 50000 tiles exactly
  const int hi = lane >> 4, m = lane & 15;
  const int rowA = tile * 16 + m;
  const int srcA = edge_index[rowA];
  const int dstA = edge_index[NE + rowA];
  const float* pS = h + srcA * HD;
  const float* pD = h + dstA * HD;
  const float* pE = e + rowA * HD;

  // GEMM1: [16x192] x [192x128]; B fragments preloaded per K-chunk so the
  // 16 b128 loads issue as one clause and overlap with WMMA execution.
  v8f acc[8] = {};
#pragma unroll
  for (int c = 0; c < 6; ++c) {
    const float* p = (c < 2) ? (pS + 32 * c) : (c < 4) ? (pD + 32 * (c - 2))
                                                       : (pE + 32 * (c - 4));
    v16bf bfrag[8];
#pragma unroll
    for (int t = 0; t < 8; ++t)
      bfrag[t] = *(const v16bf*)(W1p + (((t * 6 + c) * 32 + lane) << 4));
    v16bf a = {};
#pragma unroll
    for (int j = 0; j < 16; ++j) a[j] = (__bf16)p[kmap(j, hi)];
#pragma unroll
    for (int t = 0; t < 8; ++t) acc[t] = wmma_bf16(a, bfrag[t], acc[t]);
  }
  // bias + ReLU, transpose C/D-layout -> A-layout via LDS
  unsigned short* Tw = T[wave];
#pragma unroll
  for (int t = 0; t < 8; ++t) {
    int n = t * 16 + m;
    float bb = b1[n];
#pragma unroll
    for (int r = 0; r < 8; ++r) {
      float v = acc[t][r] + bb;
      v = v > 0.f ? v : 0.f;
      Tw[(r + (hi << 3)) * 128 + n] = f2bf(v);
    }
  }
  asm volatile("s_wait_dscnt 0" ::: "memory");
  // GEMM2: [16x128] x [128x64]
  v8f acc2[4] = {};
#pragma unroll
  for (int c = 0; c < 4; ++c) {
    v16bf bfrag[4];
#pragma unroll
    for (int t = 0; t < 4; ++t)
      bfrag[t] = *(const v16bf*)(W2p + (((t * 4 + c) * 32 + lane) << 4));
    v16bf a = {};
#pragma unroll
    for (int j = 0; j < 16; ++j) a[j] = us2bf(Tw[m * 128 + 32 * c + kmap(j, hi)]);
#pragma unroll
    for (int t = 0; t < 4; ++t) acc2[t] = wmma_bf16(a, bfrag[t], acc2[t]);
  }
  int dstR[8];
#pragma unroll
  for (int r = 0; r < 8; ++r) dstR[r] = edge_index[NE + tile * 16 + r + (hi << 3)];
#pragma unroll
  for (int t = 0; t < 4; ++t) {
    int n = t * 16 + m;
    float bb = b2[n];
#pragma unroll
    for (int r = 0; r < 8; ++r) {
      float v = acc2[t][r] + bb;
      int row = tile * 16 + r + (hi << 3);
      e[row * HD + n] = v;
      atomicAdd(&agg[dstR[r] * HD + n], v);   // segment_sum over dst
    }
  }
}

// ---- node update: h = MLP(128->128->64)(h || agg) ------------------------

__global__ void __launch_bounds__(128)
node_update_kernel(float* __restrict__ h, const float* __restrict__ agg,
                   const unsigned short* __restrict__ W1p, const float* __restrict__ b1,
                   const unsigned short* __restrict__ W2p, const float* __restrict__ b2) {
  __shared__ unsigned short T[4][16 * 128];
  const int lane = threadIdx.x & 31;
  const int wave = threadIdx.x >> 5;
  const int tile = blockIdx.x * 4 + wave;
  if (tile * 16 >= NN) return;                      // wave-uniform
  const int hi = lane >> 4, m = lane & 15;
  const int rowA = tile * 16 + m;
  const float* pH = h + rowA * HD;
  const float* pA = agg + rowA * HD;

  v8f acc[8] = {};
#pragma unroll
  for (int c = 0; c < 4; ++c) {
    const float* p = (c < 2) ? (pH + 32 * c) : (pA + 32 * (c - 2));
    v16bf bfrag[8];
#pragma unroll
    for (int t = 0; t < 8; ++t)
      bfrag[t] = *(const v16bf*)(W1p + (((t * 4 + c) * 32 + lane) << 4));
    v16bf a = {};
#pragma unroll
    for (int j = 0; j < 16; ++j) a[j] = (__bf16)p[kmap(j, hi)];
#pragma unroll
    for (int t = 0; t < 8; ++t) acc[t] = wmma_bf16(a, bfrag[t], acc[t]);
  }
  unsigned short* Tw = T[wave];
#pragma unroll
  for (int t = 0; t < 8; ++t) {
    int n = t * 16 + m;
    float bb = b1[n];
#pragma unroll
    for (int r = 0; r < 8; ++r) {
      float v = acc[t][r] + bb;
      v = v > 0.f ? v : 0.f;
      Tw[(r + (hi << 3)) * 128 + n] = f2bf(v);
    }
  }
  asm volatile("s_wait_dscnt 0" ::: "memory");
  v8f acc2[4] = {};
#pragma unroll
  for (int c = 0; c < 4; ++c) {
    v16bf bfrag[4];
#pragma unroll
    for (int t = 0; t < 4; ++t)
      bfrag[t] = *(const v16bf*)(W2p + (((t * 4 + c) * 32 + lane) << 4));
    v16bf a = {};
#pragma unroll
    for (int j = 0; j < 16; ++j) a[j] = us2bf(Tw[m * 128 + 32 * c + kmap(j, hi)]);
#pragma unroll
    for (int t = 0; t < 4; ++t) acc2[t] = wmma_bf16(a, bfrag[t], acc2[t]);
  }
#pragma unroll
  for (int t = 0; t < 4; ++t) {
    int n = t * 16 + m;
    float bb = b2[n];
#pragma unroll
    for (int r = 0; r < 8; ++r) {
      int row = tile * 16 + r + (hi << 3);
      h[row * HD + n] = acc2[t][r] + bb;
    }
  }
}

// ---- edge head: MLP(192->128->64->1) -------------------------------------

__global__ void __launch_bounds__(128)
edge_head_kernel(const float* __restrict__ h, const float* __restrict__ e,
                 const int* __restrict__ edge_index,
                 const unsigned short* __restrict__ W1p, const float* __restrict__ b1,
                 const unsigned short* __restrict__ W2p, const float* __restrict__ b2,
                 const float* __restrict__ W3, const float* __restrict__ b3,
                 float* __restrict__ out) {
  __shared__ unsigned short T[4][16 * 128];
  __shared__ float red[4][16][17];
  const int lane = threadIdx.x & 31;
  const int wave = threadIdx.x >> 5;
  const int tile = blockIdx.x * 4 + wave;
  const int hi = lane >> 4, m = lane & 15;
  const int rowA = tile * 16 + m;
  const int srcA = edge_index[rowA];
  const int dstA = edge_index[NE + rowA];
  const float* pS = h + srcA * HD;
  const float* pD = h + dstA * HD;
  const float* pE = e + rowA * HD;

  v8f acc[8] = {};
#pragma unroll
  for (int c = 0; c < 6; ++c) {
    const float* p = (c < 2) ? (pS + 32 * c) : (c < 4) ? (pD + 32 * (c - 2))
                                                       : (pE + 32 * (c - 4));
    v16bf bfrag[8];
#pragma unroll
    for (int t = 0; t < 8; ++t)
      bfrag[t] = *(const v16bf*)(W1p + (((t * 6 + c) * 32 + lane) << 4));
    v16bf a = {};
#pragma unroll
    for (int j = 0; j < 16; ++j) a[j] = (__bf16)p[kmap(j, hi)];
#pragma unroll
    for (int t = 0; t < 8; ++t) acc[t] = wmma_bf16(a, bfrag[t], acc[t]);
  }
  unsigned short* Tw = T[wave];
#pragma unroll
  for (int t = 0; t < 8; ++t) {
    int n = t * 16 + m;
    float bb = b1[n];
#pragma unroll
    for (int r = 0; r < 8; ++r) {
      float v = acc[t][r] + bb;
      v = v > 0.f ? v : 0.f;
      Tw[(r + (hi << 3)) * 128 + n] = f2bf(v);
    }
  }
  asm volatile("s_wait_dscnt 0" ::: "memory");
  v8f acc2[4] = {};
#pragma unroll
  for (int c = 0; c < 4; ++c) {
    v16bf bfrag[4];
#pragma unroll
    for (int t = 0; t < 4; ++t)
      bfrag[t] = *(const v16bf*)(W2p + (((t * 4 + c) * 32 + lane) << 4));
    v16bf a = {};
#pragma unroll
    for (int j = 0; j < 16; ++j) a[j] = us2bf(Tw[m * 128 + 32 * c + kmap(j, hi)]);
#pragma unroll
    for (int t = 0; t < 4; ++t) acc2[t] = wmma_bf16(a, bfrag[t], acc2[t]);
  }
  // ReLU + final 64->1 dot, reduced across the 16 lanes holding each row
  float part[8];
#pragma unroll
  for (int r = 0; r < 8; ++r) part[r] = 0.f;
#pragma unroll
  for (int t = 0; t < 4; ++t) {
    int n = t * 16 + m;
    float bb = b2[n];
    float w = W3[n];
#pragma unroll
    for (int r = 0; r < 8; ++r) {
      float v = acc2[t][r] + bb;
      v = v > 0.f ? v : 0.f;
      part[r] += v * w;
    }
  }
#pragma unroll
  for (int r = 0; r < 8; ++r) red[wave][r + (hi << 3)][m] = part[r];
  asm volatile("s_wait_dscnt 0" ::: "memory");
  if (lane < 16) {
    float s = b3[0];
#pragma unroll
    for (int j = 0; j < 16; ++j) s += red[wave][lane][j];
    out[tile * 16 + lane] = s;
  }
}

// ---- host launch ---------------------------------------------------------

extern "C" void kernel_launch(void* const* d_in, const int* in_sizes, int n_in,
                              void* d_out, int out_size, void* d_ws, size_t ws_size,
                              hipStream_t stream) {
  (void)out_size; (void)ws_size;
  const float* x = nullptr;
  const float* edge_attr = nullptr;
  const int* edge_index = nullptr;
  const float* P[64];
  int Psz[64];
  int np = 0;
  for (int i = 0; i < n_in; ++i) {
    int s = in_sizes[i];
    if (s == NN * 6) x = (const float*)d_in[i];
    else if (s == NE * 8) edge_attr = (const float*)d_in[i];
    else if (s == 2 * NE) edge_index = (const int*)d_in[i];
    else if (np < 64) { Psz[np] = s; P[np] = (const float*)d_in[i]; ++np; }
  }
  if (!x || !edge_attr || !edge_index || np < 38) return;

  const float *neW0, *neb0, *neW1, *neb1;
  const float *eeW0, *eeb0, *eeW1, *eeb1;
  const float *euW1[3], *eub1[3], *euW2[3], *eub2[3];
  const float *nuW1[3], *nub1[3], *nuW2[3], *nub2[3];
  const float *ehW1, *ehb1, *ehW2, *ehb2, *ehW3, *ehb3;

  int o = 0;
  if (Psz[0] == 6 * HD) {
    // insertion order: node_encoder, edge_encoder, edge_updates, node_updates, edge_head
    neW0 = P[o++]; neb0 = P[o++]; neW1 = P[o++]; neb1 = P[o++];
    eeW0 = P[o++]; eeb0 = P[o++]; eeW1 = P[o++]; eeb1 = P[o++];
    for (int k = 0; k < 3; ++k) { euW1[k] = P[o++]; eub1[k] = P[o++]; euW2[k] = P[o++]; eub2[k] = P[o++]; }
    for (int k = 0; k < 3; ++k) { nuW1[k] = P[o++]; nub1[k] = P[o++]; nuW2[k] = P[o++]; nub2[k] = P[o++]; }
    ehW1 = P[o++]; ehb1 = P[o++]; ehW2 = P[o++]; ehb2 = P[o++]; ehW3 = P[o++]; ehb3 = P[o++];
  } else {
    // sorted-key pytree order: edge_encoder, edge_head, edge_updates, node_encoder, node_updates
    eeW0 = P[o++]; eeb0 = P[o++]; eeW1 = P[o++]; eeb1 = P[o++];
    ehW1 = P[o++]; ehb1 = P[o++]; ehW2 = P[o++]; ehb2 = P[o++]; ehW3 = P[o++]; ehb3 = P[o++];
    for (int k = 0; k < 3; ++k) { euW1[k] = P[o++]; eub1[k] = P[o++]; euW2[k] = P[o++]; eub2[k] = P[o++]; }
    neW0 = P[o++]; neb0 = P[o++]; neW1 = P[o++]; neb1 = P[o++];
    for (int k = 0; k < 3; ++k) { nuW1[k] = P[o++]; nub1[k] = P[o++]; nuW2[k] = P[o++]; nub2[k] = P[o++]; }
  }

  // workspace layout
  uintptr_t base = (uintptr_t)d_ws;
  size_t off = 0;
  auto alloc = [&](size_t bytes) -> void* {
    void* p = (void*)(base + off);
    off += (bytes + 255) & ~(size_t)255;
    return p;
  };
  float* h   = (float*)alloc(sizeof(float) * NN * HD);
  float* agg = (float*)alloc(sizeof(float) * NN * HD);
  float* e   = (float*)alloc(sizeof(float) * (size_t)NE * HD);
  unsigned short *euW1p[3], *euW2p[3], *nuW1p[3], *nuW2p[3];
  for (int k = 0; k < 3; ++k) {
    euW1p[k] = (unsigned short*)alloc(2 * 192 * 128);
    euW2p[k] = (unsigned short*)alloc(2 * 128 * 64);
    nuW1p[k] = (unsigned short*)alloc(2 * 128 * 128);
    nuW2p[k] = (unsigned short*)alloc(2 * 128 * 64);
  }
  unsigned short* ehW1p = (unsigned short*)alloc(2 * 192 * 128);
  unsigned short* ehW2p = (unsigned short*)alloc(2 * 128 * 64);

  auto packW = [&](const float* W, unsigned short* outp, int K, int N) {
    int total = (K / 32) * (N / 16) * 512;
    pack_weight_kernel<<<(total + 255) / 256, 256, 0, stream>>>(W, outp, K, N);
  };
  for (int k = 0; k < 3; ++k) {
    packW(euW1[k], euW1p[k], 192, 128);
    packW(euW2[k], euW2p[k], 128, 64);
    packW(nuW1[k], nuW1p[k], 128, 128);
    packW(nuW2[k], nuW2p[k], 128, 64);
  }
  packW(ehW1, ehW1p, 192, 128);
  packW(ehW2, ehW2p, 128, 64);

  node_encoder_kernel<<<(NN + 127) / 128, 128, 0, stream>>>(x, neW0, neb0, neW1, neb1, h);
  edge_encoder_kernel<<<(NE + 127) / 128, 128, 0, stream>>>(edge_attr, eeW0, eeb0, eeW1, eeb1, e);

  for (int k = 0; k < 3; ++k) {
    zero_kernel<<<(NN * HD + 255) / 256, 256, 0, stream>>>(agg, NN * HD);
    edge_update_kernel<<<NE / 16 / 4, 128, 0, stream>>>(h, e, edge_index, agg,
                                                        euW1p[k], eub1[k], euW2p[k], eub2[k]);
    node_update_kernel<<<(NN / 16 + 3) / 4, 128, 0, stream>>>(h, agg,
                                                              nuW1p[k], nub1[k], nuW2p[k], nub2[k]);
  }
  edge_head_kernel<<<NE / 16 / 4, 128, 0, stream>>>(h, e, edge_index,
                                                    ehW1p, ehb1, ehW2p, ehb2, ehW3, ehb3,
                                                    (float*)d_out);
}
